// NodeRecommendationModel_91285234909247
// MI455X (gfx1250) — matrix operand order repaired and checked by reading, not compiled, not run on hardware
//
#include <hip/hip_runtime.h>

// ---------------------------------------------------------------------------
// GraphSAGE (2 layers + FC) for MI455X / gfx1250, wave32, WMMA bf16.
//
//   h1 = relu(feat @ Ws1 + mean_agg(feat) @ Wn1 + b1)
//   h2 =      h1   @ Ws2 + mean_agg(h1)   @ Wn2 + b2
//   out = h2 @ Wfc + bfc                        (f32, 50000 x 128)
//
// Activations kept in bf16 (halves random gather traffic; L2-resident),
// aggregation in f32 via global_atomic_add_f32, GEMMs via
// v_wmma_f32_16x16x32_bf16 with pre-swizzled weight fragments.
// ---------------------------------------------------------------------------

typedef __attribute__((ext_vector_type(16))) __bf16 v16bf;
typedef __attribute__((ext_vector_type(8)))  float  v8f;
typedef __attribute__((ext_vector_type(4)))  unsigned int u32x4;

union Frag { v16bf h; u32x4 q[2]; };

#define N_NODES 50000
#define N_EDGES 1600000
#define FEATS   128
#define NP      50048   // 391 * 128, padded row count

__device__ __forceinline__ unsigned short f2bf(float f) {
    unsigned u = __float_as_uint(f);
    unsigned r = u + 0x7fffu + ((u >> 16) & 1u);   // round-to-nearest-even
    return (unsigned short)(r >> 16);
}
__device__ __forceinline__ float bf2f(unsigned short h) {
    return __uint_as_float(((unsigned)h) << 16);
}

// --- pre-swizzle one 128x128 f32 weight into WMMA B-fragment order (bf16) ---
// Fragment (kt,nt): lane L holds col n = nt*16 + (L&15),
//   K = kt*32 + (L>>4)*16 + {0..15}, packed 2 bf16 per dword, 8 dwords/lane.
__global__ __launch_bounds__(256) void pack_weight_kernel(
    const float* __restrict__ W, unsigned int* __restrict__ P) {
    int tid  = blockIdx.x * 256 + threadIdx.x;      // 0..1023
    int lane = tid & 31;
    int nt   = (tid >> 5) & 7;
    int kt   = tid >> 8;
    int n      = nt * 16 + (lane & 15);
    int kstart = kt * 32 + (lane >> 4) * 16;
#pragma unroll
    for (int j = 0; j < 8; ++j) {
        int k = kstart + 2 * j;
        unsigned lo = f2bf(W[k * FEATS + n]);
        unsigned hi = f2bf(W[(k + 1) * FEATS + n]);
        P[tid * 8 + j] = lo | (hi << 16);
    }
}

// --- f32 -> bf16 copy with zero padding -------------------------------------
__global__ __launch_bounds__(256) void cvt_kernel(
    const float* __restrict__ in, unsigned short* __restrict__ out,
    int nValid, int nTotal) {
    int i = blockIdx.x * 256 + threadIdx.x;
    if (i < nTotal) out[i] = (i < nValid) ? f2bf(in[i]) : (unsigned short)0;
}

// --- zero agg + deg ---------------------------------------------------------
__global__ __launch_bounds__(256) void zero_kernel(
    float* __restrict__ agg, float* __restrict__ deg, int nAgg, int nDeg) {
    int i = blockIdx.x * 256 + threadIdx.x;
    if (i < nAgg) agg[i] = 0.0f;
    if (i < nDeg) deg[i] = 0.0f;
}

// --- in-degree (computed once; identical for both layers) -------------------
__global__ __launch_bounds__(256) void deg_kernel(
    const int* __restrict__ dst, float* __restrict__ deg, int E) {
    int e = blockIdx.x * 256 + threadIdx.x;
    if (e < E) atomicAdd(&deg[dst[e]], 1.0f);
}

// --- edge gather + scatter-add: one wave per edge, 4 feats per lane ---------
__global__ __launch_bounds__(256) void scatter_kernel(
    const unsigned short* __restrict__ h, const int* __restrict__ src,
    const int* __restrict__ dst, float* __restrict__ agg, int E) {
    long long tid = (long long)blockIdx.x * 256 + threadIdx.x;
    int e = (int)(tid >> 5);
    int c = (int)(tid & 31);
    if (e >= E) return;
    int s = src[e];   // wave-uniform -> scalarized
    int d = dst[e];
    unsigned long long v =
        *(const unsigned long long*)(h + s * FEATS + c * 4);   // 4 bf16
    float* out = agg + (long long)d * FEATS + c * 4;
    atomicAdd(out + 0, bf2f((unsigned short)(v)));
    atomicAdd(out + 1, bf2f((unsigned short)(v >> 16)));
    atomicAdd(out + 2, bf2f((unsigned short)(v >> 32)));
    atomicAdd(out + 3, bf2f((unsigned short)(v >> 48)));
}

// --- mean (divide by clamped degree), cvt to bf16, and re-zero agg ----------
__global__ __launch_bounds__(256) void divcvt_kernel(
    float* __restrict__ agg, const float* __restrict__ deg,
    unsigned short* __restrict__ out, int nValid, int nTotal) {
    int i = blockIdx.x * 256 + threadIdx.x;
    if (i >= nTotal) return;
    if (i < nValid) {
        float dv = fmaxf(deg[i >> 7], 1.0f);
        out[i] = f2bf(agg[i] / dv);
    } else {
        out[i] = 0;
    }
    agg[i] = 0.0f;   // ready for next layer's scatter
}

// --- fused GEMM: C = act( A1@W1 [+ A2@W2] + bias ) --------------------------
// One wave per 16-row x 128-col strip; 8 n-tiles x 4 k-steps of bf16 WMMA.
template <bool HAS2>
__global__ __launch_bounds__(256) void gemm_kernel(
    const unsigned short* __restrict__ A1, const unsigned int* __restrict__ W1,
    const unsigned short* __restrict__ A2, const unsigned int* __restrict__ W2,
    const float* __restrict__ bias,
    unsigned short* __restrict__ outB,      // bf16 out (padded) or null
    float* __restrict__ outF,               // f32 out (row-guarded) or null
    int nRows, int relu) {
    const int lane    = threadIdx.x & 31;
    const int wave    = threadIdx.x >> 5;
    const int rowBase = blockIdx.x * 128 + wave * 16;
    const int hi      = lane >> 4;
    const int lo      = lane & 15;
    const int arow    = rowBase + lo;

    v8f acc[8] = {};

    for (int kt = 0; kt < 4; ++kt) {
        const int kb = kt * 32 + hi * 16;
        Frag a1, a2;
        a1.q[0] = *(const u32x4*)(A1 + arow * FEATS + kb);
        a1.q[1] = *(const u32x4*)(A1 + arow * FEATS + kb + 8);
        if (HAS2) {
            a2.q[0] = *(const u32x4*)(A2 + arow * FEATS + kb);
            a2.q[1] = *(const u32x4*)(A2 + arow * FEATS + kb + 8);
        }
#pragma unroll
        for (int nt = 0; nt < 8; ++nt) {
            Frag b1;
            const u32x4* pw = (const u32x4*)(W1 + ((kt * 8 + nt) * 32 + lane) * 8);
            b1.q[0] = pw[0];
            b1.q[1] = pw[1];
            acc[nt] = __builtin_amdgcn_wmma_f32_16x16x32_bf16(
                false, a1.h, false, b1.h, (short)0, acc[nt], false, false);
            if (HAS2) {
                Frag b2;
                const u32x4* pw2 =
                    (const u32x4*)(W2 + ((kt * 8 + nt) * 32 + lane) * 8);
                b2.q[0] = pw2[0];
                b2.q[1] = pw2[1];
                acc[nt] = __builtin_amdgcn_wmma_f32_16x16x32_bf16(
                    false, a2.h, false, b2.h, (short)0, acc[nt], false, false);
            }
        }
    }

    // epilogue: bias (+relu), store bf16 and/or guarded f32
#pragma unroll
    for (int nt = 0; nt < 8; ++nt) {
        const int col = nt * 16 + lo;
        const float bv = bias[col];
#pragma unroll
        for (int v = 0; v < 8; ++v) {
            const int row = rowBase + hi * 8 + v;
            float x = acc[nt][v] + bv;
            if (relu) x = fmaxf(x, 0.0f);
            if (outB) outB[row * FEATS + col] = f2bf(x);
            if (outF && row < nRows) outF[row * FEATS + col] = x;
        }
    }
}

extern "C" void kernel_launch(void* const* d_in, const int* in_sizes, int n_in,
                              void* d_out, int out_size, void* d_ws,
                              size_t ws_size, hipStream_t stream) {
    const float* feat  = (const float*)d_in[0];
    const int*   src   = (const int*)d_in[1];
    const int*   dst   = (const int*)d_in[2];
    const float* Ws1   = (const float*)d_in[3];
    const float* Wn1   = (const float*)d_in[4];
    const float* b1    = (const float*)d_in[5];
    const float* Ws2   = (const float*)d_in[6];
    const float* Wn2   = (const float*)d_in[7];
    const float* b2    = (const float*)d_in[8];
    const float* Wfc   = (const float*)d_in[9];
    const float* bfc   = (const float*)d_in[10];
    float*       out   = (float*)d_out;
    (void)in_sizes; (void)n_in; (void)out_size; (void)ws_size;

    // workspace carve-up (256B aligned)
    char*  ws  = (char*)d_ws;
    size_t off = 0;
    auto alloc = [&](size_t bytes) -> void* {
        void* p = ws + off;
        off += (bytes + 255) & ~(size_t)255;
        return p;
    };
    float*          agg   = (float*)alloc((size_t)NP * FEATS * 4);
    float*          deg   = (float*)alloc((size_t)NP * 4);
    unsigned short* featB = (unsigned short*)alloc((size_t)NP * FEATS * 2);
    unsigned short* h1b   = (unsigned short*)alloc((size_t)NP * FEATS * 2);
    unsigned short* aggB  = (unsigned short*)alloc((size_t)NP * FEATS * 2);
    unsigned short* h2b   = (unsigned short*)alloc((size_t)NP * FEATS * 2);
    unsigned int*   Wp    = (unsigned int*)alloc((size_t)5 * 8192 * 4);
    unsigned int* Wp_s1 = Wp;
    unsigned int* Wp_n1 = Wp + 8192;
    unsigned int* Wp_s2 = Wp + 16384;
    unsigned int* Wp_n2 = Wp + 24576;
    unsigned int* Wp_fc = Wp + 32768;

    const int elemBlocks = (NP * FEATS + 255) / 256;
    const int edgeBlocks = (N_EDGES + 255) / 256;
    const long long scatterThreads = (long long)N_EDGES * 32;
    const int scatterBlocks = (int)((scatterThreads + 255) / 256);
    const int gemmBlocks = NP / 128;   // 391

    // weight pre-swizzle
    pack_weight_kernel<<<4, 256, 0, stream>>>(Ws1, Wp_s1);
    pack_weight_kernel<<<4, 256, 0, stream>>>(Wn1, Wp_n1);
    pack_weight_kernel<<<4, 256, 0, stream>>>(Ws2, Wp_s2);
    pack_weight_kernel<<<4, 256, 0, stream>>>(Wn2, Wp_n2);
    pack_weight_kernel<<<4, 256, 0, stream>>>(Wfc, Wp_fc);

    // features -> bf16 (zero pad rows); zero accumulators; degree (once)
    cvt_kernel<<<elemBlocks, 256, 0, stream>>>(feat, featB, N_NODES * FEATS,
                                               NP * FEATS);
    zero_kernel<<<elemBlocks, 256, 0, stream>>>(agg, deg, NP * FEATS, NP);
    deg_kernel<<<edgeBlocks, 256, 0, stream>>>(dst, deg, N_EDGES);

    // layer 1
    scatter_kernel<<<scatterBlocks, 256, 0, stream>>>(featB, src, dst, agg,
                                                      N_EDGES);
    divcvt_kernel<<<elemBlocks, 256, 0, stream>>>(agg, deg, aggB,
                                                  N_NODES * FEATS, NP * FEATS);
    gemm_kernel<true><<<gemmBlocks, 256, 0, stream>>>(
        featB, Wp_s1, aggB, Wp_n1, b1, h1b, nullptr, N_NODES, 1);

    // layer 2
    scatter_kernel<<<scatterBlocks, 256, 0, stream>>>(h1b, src, dst, agg,
                                                      N_EDGES);
    divcvt_kernel<<<elemBlocks, 256, 0, stream>>>(agg, deg, aggB,
                                                  N_NODES * FEATS, NP * FEATS);
    gemm_kernel<true><<<gemmBlocks, 256, 0, stream>>>(
        h1b, Wp_s2, aggB, Wp_n2, b2, h2b, nullptr, N_NODES, 0);

    // final FC -> f32 output (row-guarded)
    gemm_kernel<false><<<gemmBlocks, 256, 0, stream>>>(
        h2b, Wp_fc, nullptr, nullptr, bfc, nullptr, out, N_NODES, 0);
}
